// ScaledDotProductAttention_22170621182007
// MI455X (gfx1250) — matrix operand order repaired and checked by reading, not compiled, not run on hardware
//
#include <hip/hip_runtime.h>
#include <math.h>

// Problem constants (from reference): B=4, S=4096, D=1024, DK=DV=64
#define BATCH 4
#define SEQ   4096
#define DIM   1024
#define DKV   64
#define QT    (SEQ / 16)   // 256 row/col tiles per sequence

typedef __attribute__((ext_vector_type(16))) __bf16 v16bf;
typedef __attribute__((ext_vector_type(8)))  __bf16 v8bf;
typedef __attribute__((ext_vector_type(8)))  float  v8f;

#define WMMA_BF16(a, b, c) \
  __builtin_amdgcn_wmma_f32_16x16x32_bf16(false, (a), false, (b), (short)0, (c), false, false)

static __device__ __forceinline__ v16bf cat8(v8bf lo, v8bf hi) {
  return __builtin_shufflevector(lo, hi, 0, 1, 2, 3, 4, 5, 6, 7,
                                         8, 9, 10, 11, 12, 13, 14, 15);
}

// ---------------- WMMA fragment loaders (ISA §7.12.2 layouts, wave32) ---------------
// bf16 loaders issue two 16-byte loads per lane, the f32 loader four (16B-aligned by construction).

// A (16x32 bf16), row-major source, leading dim ld (multiple of 8).
// Lane m = lane&15; element e -> k = (e<8?0:16) + 8*half + (e&7)  == two contiguous 8-groups.
static __device__ __forceinline__ v16bf load_a_bf16(const __bf16* __restrict__ src, int ld) {
  const int lane = threadIdx.x & 31;
  const int m = lane & 15, h = lane >> 4;
  const __bf16* p = src + (size_t)m * ld + h * 8;
  const v8bf lo = *(const v8bf*)(p);        // k = 8h + 0..7
  const v8bf hi = *(const v8bf*)(p + 16);   // k = 16 + 8h + 0..7
  return cat8(lo, hi);
}

// A (16x32) from f32 source with on-the-fly bf16 conversion (two pairs of float4).
static __device__ __forceinline__ v16bf load_a_f32(const float* __restrict__ src, int ld) {
  const int lane = threadIdx.x & 31;
  const int m = lane & 15, h = lane >> 4;
  const float* p = src + (size_t)m * ld + h * 8;
  const float4 a0 = *(const float4*)(p);
  const float4 a1 = *(const float4*)(p + 4);
  const float4 b0 = *(const float4*)(p + 16);
  const float4 b1 = *(const float4*)(p + 20);
  v16bf a;
  a[0] = (__bf16)a0.x;  a[1] = (__bf16)a0.y;  a[2] = (__bf16)a0.z;  a[3] = (__bf16)a0.w;
  a[4] = (__bf16)a1.x;  a[5] = (__bf16)a1.y;  a[6] = (__bf16)a1.z;  a[7] = (__bf16)a1.w;
  a[8] = (__bf16)b0.x;  a[9] = (__bf16)b0.y;  a[10] = (__bf16)b0.z; a[11] = (__bf16)b0.w;
  a[12] = (__bf16)b1.x; a[13] = (__bf16)b1.y; a[14] = (__bf16)b1.z; a[15] = (__bf16)b1.w;
  return a;
}

// B (32x16) as the TRANSPOSE of a row-major [N x K] bf16 source:
// element (k, n) = src[n*ld + k]; lane n holds k = 8h + 0..7 and 16 + 8h + 0..7.
static __device__ __forceinline__ v16bf load_bT_bf16(const __bf16* __restrict__ src, int ld) {
  const int lane = threadIdx.x & 31;
  const int n = lane & 15, h = lane >> 4;
  const __bf16* p = src + (size_t)n * ld + h * 8;
  const v8bf lo = *(const v8bf*)(p);
  const v8bf hi = *(const v8bf*)(p + 16);
  return cat8(lo, hi);
}

// ---------------- Pass 0: pack weights W(DxN f32) -> Wt(NxD bf16) ---------------

__global__ void __launch_bounds__(256)
wpack_kernel(const float* __restrict__ Wq, const float* __restrict__ Wk,
             const float* __restrict__ Wv, __bf16* __restrict__ Wt) {
  // Wt flat layout: [which][n (DKV)][k (DIM)]
  const size_t i = (size_t)blockIdx.x * blockDim.x + threadIdx.x;  // 3*DKV*DIM
  const int k = (int)(i & (DIM - 1));
  const int n = (int)((i >> 10) & (DKV - 1));
  const int which = (int)(i >> 16);             // DKV*DIM = 2^16
  const float* W = (which == 0) ? Wq : ((which == 1) ? Wk : Wv);
  Wt[i] = (__bf16)W[(size_t)k * DKV + n];
}

// ---------------- Pass 1: Q/K/V projections (bf16; V stored transposed) ---------------

__global__ void __launch_bounds__(128)
proj_kernel(const float* __restrict__ x, const __bf16* __restrict__ Wt,
            __bf16* __restrict__ Qb, __bf16* __restrict__ Kb, __bf16* __restrict__ VT) {
  const int tile  = blockIdx.x;                 // over B*S/16 row tiles
  const int which = blockIdx.y;                 // 0:Q 1:K 2:V

  const int wave = threadIdx.x >> 5;            // 4 waves -> 4 column tiles of 16
  const int n0   = wave * 16;
  const int row0 = tile * 16;
  const __bf16* Wn = Wt + ((size_t)which * DKV + n0) * DIM;

  v8f c = {};
  for (int kk = 0; kk < DIM; kk += 32) {
    if (kk + 32 < DIM)
      __builtin_prefetch(x + (size_t)row0 * DIM + kk + 32, 0, 0);
    v16bf a = load_a_f32(x + (size_t)row0 * DIM + kk, DIM);
    v16bf b = load_bT_bf16(Wn + kk, DIM);       // contiguous: Wt[(n0+n)*DIM + kk + k]
    c = WMMA_BF16(a, b, c);
  }

  const int lane = threadIdx.x & 31;
  const int col = lane & 15, h = lane >> 4;
  if (which == 2) {
    // V transposed: VT[(b*64 + col)*SEQ + s]; this lane's 8 rows are contiguous in s.
    const int bidx = row0 >> 12;                // row0 / SEQ
    const int s    = row0 & (SEQ - 1);
    v8bf pk;
#pragma unroll
    for (int r = 0; r < 8; ++r) pk[r] = (__bf16)c[r];
    *(v8bf*)(VT + (size_t)(bidx * DKV + col) * SEQ + s + 8 * h) = pk;
  } else {
    __bf16* dst = (which == 0) ? Qb : Kb;
#pragma unroll
    for (int r = 0; r < 8; ++r)
      dst[(size_t)(row0 + r + 8 * h) * DKV + n0 + col] = (__bf16)c[r];
  }
}

// ---------------- Pass 2: per-key-column softmax stats over q >= s ---------------

__global__ void __launch_bounds__(32)
colstats_kernel(const __bf16* __restrict__ Qb, const __bf16* __restrict__ Kb,
                float* __restrict__ mcol, float* __restrict__ lcol) {
  const int b  = blockIdx.x >> 8;               // QT == 256
  const int st = blockIdx.x & (QT - 1);
  const int s0 = st * 16;
  const __bf16* Qbase = Qb + (size_t)b * SEQ * DKV;
  const __bf16* Kbase = Kb + (size_t)b * SEQ * DKV;

  // Fixed B operands: K^T chunks for this 16-column strip (k = 0..31, 32..63)
  const v16bf bk0 = load_bT_bf16(Kbase + (size_t)s0 * DKV + 0, DKV);
  const v16bf bk1 = load_bT_bf16(Kbase + (size_t)s0 * DKV + 32, DKV);

  const int lane = threadIdx.x & 31;
  const int scol = lane & 15, h = lane >> 4;
  const float scale = 0.125f;                   // 1/sqrt(64)

  float m_run = -INFINITY, l_run = 0.f;
  for (int qt = st; qt < QT; ++qt) {
    if (qt + 1 < QT)
      __builtin_prefetch(Qbase + (size_t)(qt + 1) * 16 * DKV, 0, 0);

    v8f c = {};
    v16bf a0 = load_a_bf16(Qbase + (size_t)qt * 16 * DKV + 0, DKV);
    v16bf a1 = load_a_bf16(Qbase + (size_t)qt * 16 * DKV + 32, DKV);
    c = WMMA_BF16(a0, bk0, c);
    c = WMMA_BF16(a1, bk1, c);

    // Each lane owns 8 q-rows of key-column (s0+scol).
    float vals[8];
    float tmax = -INFINITY;
#pragma unroll
    for (int r = 0; r < 8; ++r) {
      const int q = qt * 16 + r + 8 * h;
      const bool valid = (q >= s0 + scol);
      const float v = valid ? c[r] * scale : -INFINITY;
      vals[r] = v;
      tmax = fmaxf(tmax, v);
    }
    if (tmax > -INFINITY) {
      float tsum = 0.f;
#pragma unroll
      for (int r = 0; r < 8; ++r)
        tsum += (vals[r] > -INFINITY) ? __expf(vals[r] - tmax) : 0.f;
      const float m_new = fmaxf(m_run, tmax);
      l_run = l_run * __expf(m_run - m_new) + tsum * __expf(tmax - m_new);
      m_run = m_new;
    }
  }

  // Merge the two half-wave row slices of each column.
  const float m_o = __shfl_xor(m_run, 16, 32);
  const float l_o = __shfl_xor(l_run, 16, 32);
  const float m_f = fmaxf(m_run, m_o);
  const float l_f = ((l_run > 0.f) ? l_run * __expf(m_run - m_f) : 0.f) +
                    ((l_o  > 0.f) ? l_o  * __expf(m_o  - m_f) : 0.f);
  if (lane < 16) {
    mcol[(size_t)b * SEQ + s0 + scol] = m_f;
    lcol[(size_t)b * SEQ + s0 + scol] = l_f;
  }
}

// ---------------- Pass 2.5: VtT = VT / l[s] (coalesced elementwise) ---------------

__global__ void __launch_bounds__(256)
vt_scale_kernel(const __bf16* __restrict__ VT, const float* __restrict__ lcol,
                __bf16* __restrict__ VtT) {
  const size_t i = (size_t)blockIdx.x * blockDim.x + threadIdx.x; // over B*64*SEQ
  const int s = (int)(i & (SEQ - 1));
  const int b = (int)(i >> 18);                 // 64 * 4096 = 2^18
  VtT[i] = (__bf16)((float)VT[i] / lcol[(size_t)b * SEQ + s]);
}

// ---------------- Pass 3: out = exp(score - m[s]) @ (V / l[s]) ---------------

__global__ void __launch_bounds__(32)
attn_out_kernel(const __bf16* __restrict__ Qb, const __bf16* __restrict__ Kb,
                const __bf16* __restrict__ VtT,
                const float* __restrict__ mcol, float* __restrict__ out) {
  const int b  = blockIdx.x >> 8;
  const int qt = blockIdx.x & (QT - 1);
  const int q0 = qt * 16;
  const __bf16* Qbase = Qb + (size_t)b * SEQ * DKV;
  const __bf16* Kbase = Kb + (size_t)b * SEQ * DKV;
  const __bf16* Vtb   = VtT + (size_t)b * DKV * SEQ;   // [col][s] layout
  const float* mb = mcol + (size_t)b * SEQ;

  const int lane = threadIdx.x & 31;
  const int nl = lane & 15, h = lane >> 4;
  const float scale = 0.125f;

  // Fixed B operands: Q^T chunks for this 16-query tile.
  const v16bf bq0 = load_bT_bf16(Qbase + (size_t)q0 * DKV + 0, DKV);
  const v16bf bq1 = load_bT_bf16(Qbase + (size_t)q0 * DKV + 32, DKV);

  v8f acc[4] = {};                              // 16 q-rows x 64 out cols

  const int q = q0 + nl;                        // this lane's query column in T
  const int pairs = (qt + 2) >> 1;              // s-tile pairs covering s <= q
  for (int sp = 0; sp < pairs; ++sp) {
    const int t0 = 2 * sp, t1 = 2 * sp + 1;
    if (sp + 1 < pairs)
      __builtin_prefetch(Kbase + (size_t)(t0 + 2) * 16 * DKV, 0, 0);

    // Transposed score tile T = K_tile @ Q_tile^T: C-layout of T is exactly
    // the A-layout slice of P = exp(score), so no cross-lane transpose needed.
    v8f c0 = {};
    {
      v16bf a0 = load_a_bf16(Kbase + (size_t)t0 * 16 * DKV + 0, DKV);
      v16bf a1 = load_a_bf16(Kbase + (size_t)t0 * 16 * DKV + 32, DKV);
      c0 = WMMA_BF16(a0, bq0, c0);
      c0 = WMMA_BF16(a1, bq1, c0);
    }
    v8f c1 = {};
    if (t1 <= qt) {                             // wave-uniform branch: EXEC unchanged
      v16bf a0 = load_a_bf16(Kbase + (size_t)t1 * 16 * DKV + 0, DKV);
      v16bf a1 = load_a_bf16(Kbase + (size_t)t1 * 16 * DKV + 32, DKV);
      c1 = WMMA_BF16(a0, bq0, c1);
      c1 = WMMA_BF16(a1, bq1, c1);
    }

    // Column maxima for this lane's 8 s-rows: contiguous -> float4 pairs.
    const float4 m0a = *(const float4*)(mb + t0 * 16 + 8 * h);
    const float4 m0b = *(const float4*)(mb + t0 * 16 + 8 * h + 4);
    const float4 m1a = *(const float4*)(mb + t1 * 16 + 8 * h);
    const float4 m1b = *(const float4*)(mb + t1 * 16 + 8 * h + 4);
    const float mr0[8] = {m0a.x, m0a.y, m0a.z, m0a.w, m0b.x, m0b.y, m0b.z, m0b.w};
    const float mr1[8] = {m1a.x, m1a.y, m1a.z, m1a.w, m1b.x, m1b.y, m1b.z, m1b.w};

    // Build bf16 A-fragment of P (16 q-rows x 32 s-cols) with mask + exp.
    v16bf pa;
#pragma unroll
    for (int r = 0; r < 8; ++r) {
      const int s = t0 * 16 + r + 8 * h;
      const float p = (q >= s) ? __expf(c0[r] * scale - mr0[r]) : 0.f;
      pa[r] = (__bf16)p;
    }
#pragma unroll
    for (int r = 0; r < 8; ++r) {
      const int s = t1 * 16 + r + 8 * h;
      const float p = ((t1 <= qt) && (q >= s)) ? __expf(c1[r] * scale - mr1[r]) : 0.f;
      pa[8 + r] = (__bf16)p;
    }

    // out += P @ Vt, four 16-column tiles; Vt is [col][s] so the B-fragment is
    // the same contiguous transposed load with ld = SEQ.
    const int sbase = sp * 32;
#pragma unroll
    for (int w = 0; w < 4; ++w) {
      v16bf bv = load_bT_bf16(Vtb + (size_t)(w * 16) * SEQ + sbase, SEQ);
      acc[w] = WMMA_BF16(pa, bv, acc[w]);
    }
  }

#pragma unroll
  for (int w = 0; w < 4; ++w)
#pragma unroll
    for (int r = 0; r < 8; ++r)
      out[((size_t)b * SEQ + q0 + r + 8 * h) * DKV + w * 16 + nl] = acc[w][r];
}

// ---------------- Host launcher ---------------

extern "C" void kernel_launch(void* const* d_in, const int* in_sizes, int n_in,
                              void* d_out, int out_size, void* d_ws, size_t ws_size,
                              hipStream_t stream) {
  const float* x  = (const float*)d_in[0];
  const float* Wq = (const float*)d_in[1];
  const float* Wk = (const float*)d_in[2];
  const float* Wv = (const float*)d_in[3];
  float* out = (float*)d_out;

  // Workspace: Q, K (row-major bf16), VT & VtT (col-major bf16), m/l (f32),
  // packed weights Wt (bf16)  -> ~8.7 MB
  const size_t nqkv = (size_t)BATCH * SEQ * DKV;
  __bf16* Qb  = (__bf16*)d_ws;
  __bf16* Kb  = Qb + nqkv;
  __bf16* VT  = Kb + nqkv;
  __bf16* VtT = VT + nqkv;
  float* mcol = (float*)(VtT + nqkv);
  float* lcol = mcol + (size_t)BATCH * SEQ;
  __bf16* Wt  = (__bf16*)(lcol + (size_t)BATCH * SEQ);

  wpack_kernel<<<(3 * DKV * DIM) / 256, 256, 0, stream>>>(Wq, Wk, Wv, Wt);
  dim3 gproj(BATCH * SEQ / 16, 3);
  proj_kernel<<<gproj, 128, 0, stream>>>(x, Wt, Qb, Kb, VT);
  colstats_kernel<<<BATCH * QT, 32, 0, stream>>>(Qb, Kb, mcol, lcol);
  vt_scale_kernel<<<(BATCH * DKV * SEQ) / 256, 256, 0, stream>>>(VT, lcol, VtT);
  attn_out_kernel<<<BATCH * QT, 32, 0, stream>>>(Qb, Kb, VtT, mcol, out);
}
